// KANLayer_71193377898848
// MI455X (gfx1250) — matrix operand order repaired and checked by reading, not compiled, not run on hardware
//
#include <hip/hip_runtime.h>
#include <hip/hip_bf16.h>

// ---------------------------------------------------------------------------
// KAN layer as GEMM:  out = scale * F(8192x1152,bf16) @ G(1152x128,bf16)
//   F = [silu(x) | 8 B-spline basis values] per input dim, WMMA-A swizzled
//   G = [base_weight | spline_coeff] interleave, WMMA-B swizzled
// GEMM: v_wmma_f32_16x16x32_bf16, async global->LDS double-buffered B,
// raw split barriers (no implicit vmem fence), B reads software-pipelined.
// ---------------------------------------------------------------------------

#define IN_DIM    128
#define OUT_DIM   128
#define BATCH     8192
#define NFEAT     9                  // silu + 8 basis values
#define KDIM      (IN_DIM * NFEAT)   // 1152
#define KTILES    (KDIM / 32)        // 36
#define NTILES    (OUT_DIM / 16)     // 8

typedef __attribute__((ext_vector_type(16))) __bf16 v16bf;
typedef __attribute__((ext_vector_type(8)))  float  v8f;
typedef int v4i __attribute__((vector_size(16)));   // matches builtin param type

static __device__ __forceinline__ unsigned short f2bf(float f) {
    unsigned u = __builtin_bit_cast(unsigned, f);
    u += 0x7FFFu + ((u >> 16) & 1u);          // RNE truncation to bf16
    return (unsigned short)(u >> 16);
}

// ---- CDNA5 async global->LDS copy (ASYNCcnt) with safe fallback ----------
#if __has_builtin(__builtin_amdgcn_global_load_async_to_lds_b128)
#define KAN_ASYNC_LDS 1
static __device__ __forceinline__ void async_cp16(const void* g, void* l) {
    __builtin_amdgcn_global_load_async_to_lds_b128((v4i*)g, (v4i*)l, 0, 0);
}
static __device__ __forceinline__ void wait_async_le1() {
#if __has_builtin(__builtin_amdgcn_s_wait_asynccnt)
    __builtin_amdgcn_s_wait_asynccnt(1);
#else
    asm volatile("s_wait_asynccnt 0x1" ::: "memory");
#endif
}
static __device__ __forceinline__ void wait_async_le0() {
#if __has_builtin(__builtin_amdgcn_s_wait_asynccnt)
    __builtin_amdgcn_s_wait_asynccnt(0);
#else
    asm volatile("s_wait_asynccnt 0x0" ::: "memory");
#endif
}
// Split workgroup barrier WITHOUT the implicit vmem-drain fence of
// __syncthreads(): ordering is provided by s_wait_asynccnt (producer side)
// and by WMMA consumption of all ds_load results (consumer side).  The
// "memory" clobber is a compiler-only fence so LDS ops can't be reordered
// across it; no hardware waitcnts are generated.
static __device__ __forceinline__ void wg_barrier() {
    asm volatile("s_barrier_signal -1\n\ts_barrier_wait -1" ::: "memory");
}
#endif

// ---------------------------------------------------------------------------
// Kernel 1: silu + order-3 B-spline basis -> A-fragment swizzled bf16.
//   slot = ((rowtile*KTILES + kt)*32 + lane)*16 + v
//   lane = (b%16) | (((krel>>3)&1)<<4) ; v = (krel&7) | ((krel>>4)<<3)
// ---------------------------------------------------------------------------
__global__ __launch_bounds__(256) void kan_features(
        const float* __restrict__ x, const float* __restrict__ grid,
        unsigned short* __restrict__ Apk) {
    int idx = blockIdx.x * 256 + threadIdx.x;     // 8192*128 threads
    int b = idx & (BATCH - 1);
    int i = idx >> 13;

    float xv = x[(size_t)b * IN_DIM + i];

    float g[12];
#pragma unroll
    for (int j = 0; j < 12; ++j) g[j] = grid[i * 12 + j];

    float b0[11];
#pragma unroll
    for (int j = 0; j < 11; ++j)
        b0[j] = (xv >= g[j] && xv < g[j + 1]) ? 1.0f : 0.0f;

    float h  = g[1] - g[0];                        // uniform knots
    float r1 = __frcp_rn(h);
    float r2 = 0.5f * r1;
    float r3 = r1 * (1.0f / 3.0f);

    float b1[10];
#pragma unroll
    for (int j = 0; j < 10; ++j)
        b1[j] = ((xv - g[j]) * b0[j] + (g[j + 2] - xv) * b0[j + 1]) * r1;
    float b2[9];
#pragma unroll
    for (int j = 0; j < 9; ++j)
        b2[j] = ((xv - g[j]) * b1[j] + (g[j + 3] - xv) * b1[j + 1]) * r2;
    float b3[8];
#pragma unroll
    for (int j = 0; j < 8; ++j)
        b3[j] = ((xv - g[j]) * b2[j] + (g[j + 4] - xv) * b2[j + 1]) * r3;

    float silu = xv / (1.0f + __expf(-xv));
    float vals[NFEAT] = {silu, b3[0], b3[1], b3[2], b3[3], b3[4], b3[5], b3[6], b3[7]};

    size_t rowtile = (size_t)(b >> 4);
    int    r       = b & 15;
#pragma unroll
    for (int t = 0; t < NFEAT; ++t) {
        int k    = i * NFEAT + t;
        int kt   = k >> 5;
        int krel = k & 31;
        int lane = r | (((krel >> 3) & 1) << 4);
        int v    = (krel & 7) | ((krel >> 4) << 3);
        Apk[(((rowtile * KTILES) + kt) * 32 + lane) * 16 + v] = f2bf(vals[t]);
    }
}

// ---------------------------------------------------------------------------
// Kernel 2: pack G into WMMA B-fragment layout (16-bit B 32x16).
//   flat = ((nt*KTILES + kt)*32 + lane)*16 + p
// ---------------------------------------------------------------------------
__global__ __launch_bounds__(256) void kan_pack_w(
        const float* __restrict__ spline_coeff, const float* __restrict__ base_weight,
        unsigned short* __restrict__ Gpk) {
    int idx  = blockIdx.x * 256 + threadIdx.x;    // 8*36*32*16 = 147456
    int p    = idx & 15;
    int lane = (idx >> 4) & 31;
    int tt   = idx >> 9;
    int kt   = tt % KTILES;
    int nt   = tt / KTILES;

    int k = kt * 32 + (p & 7) + ((lane >> 4) << 3) + ((p >> 3) << 4);
    int n = nt * 16 + (lane & 15);
    int i = k / NFEAT;
    int t = k - i * NFEAT;

    float val = (t == 0) ? base_weight[i * OUT_DIM + n]
                         : spline_coeff[((size_t)(i * OUT_DIM + n) << 3) + (t - 1)];
    Gpk[idx] = f2bf(val);
}

// ---------------------------------------------------------------------------
// Kernel 3: WMMA GEMM, 8 waves/block.
//   block = 8 rowtiles (wave w -> rowtile) x 4 ntiles (blockIdx.y)
//   B staged per block into LDS (4KB/step, one async b128 per thread),
//   double buffered; A prefetched one K-step ahead through the barriers;
//   B fragment reads pipelined one WMMA ahead (2 live fragments).
// ---------------------------------------------------------------------------
__global__ __launch_bounds__(256) void kan_gemm(
        const unsigned short* __restrict__ A, const unsigned short* __restrict__ G,
        float* __restrict__ out, const float* __restrict__ scale) {
    __shared__ __align__(16) unsigned char smem[2 * 4096];

    int tid  = threadIdx.x;
    int wave = tid >> 5;
    int lane = tid & 31;
    int rowtile = blockIdx.x * 8 + wave;          // 0..511
    int nh      = blockIdx.y * 4;                 // first of 4 N-tiles

    const v16bf* Af = reinterpret_cast<const v16bf*>(A) +
                      (size_t)rowtile * KTILES * 32 + lane;

    // this thread's 16B staging slice: q = ntile-in-group, inr = 16B chunk
    int q   = tid >> 6;
    int inr = tid & 63;
    const unsigned char* gsrc = (const unsigned char*)G +
        ((size_t)(nh + q) * KTILES) * 1024 + (size_t)inr * 16;
    unsigned char* ldst0 = smem + tid * 16;
    unsigned char* ldst1 = smem + 4096 + tid * 16;

    v8f acc[4] = {};

#ifdef KAN_ASYNC_LDS
    // prologue: stage kt=0 -> buf0, kt=1 -> buf1
    async_cp16(gsrc,          ldst0);
    async_cp16(gsrc + 1024,   ldst1);
    v16bf a_cur = Af[0];

    for (int kt = 0; kt < KTILES; ++kt) {
        if (kt < KTILES - 1) wait_async_le1(); else wait_async_le0();
        wg_barrier();                              // stage kt visible to all

        // prefetch next A fragment first so it spans the whole iteration
        v16bf a_use = a_cur;
        if (kt < KTILES - 1) a_cur = Af[(size_t)(kt + 1) * 32];

        const v16bf* Bl = reinterpret_cast<const v16bf*>(smem + (kt & 1) * 4096);
        // B reads pipelined one WMMA ahead: only the first LDS round trip
        // is exposed, the rest hides under WMMA execution.
        v16bf bbA = Bl[0 * 32 + lane];
        v16bf bbB = Bl[1 * 32 + lane];
        acc[0] = __builtin_amdgcn_wmma_f32_16x16x32_bf16(false, a_use, false, bbA,
                                                         (short)0, acc[0], false, false);
        bbA = Bl[2 * 32 + lane];
        acc[1] = __builtin_amdgcn_wmma_f32_16x16x32_bf16(false, a_use, false, bbB,
                                                         (short)0, acc[1], false, false);
        bbB = Bl[3 * 32 + lane];
        acc[2] = __builtin_amdgcn_wmma_f32_16x16x32_bf16(false, a_use, false, bbA,
                                                         (short)0, acc[2], false, false);
        acc[3] = __builtin_amdgcn_wmma_f32_16x16x32_bf16(false, a_use, false, bbB,
                                                         (short)0, acc[3], false, false);

        wg_barrier();                              // buf (kt&1) free to refill
        if (kt + 2 < KTILES)
            async_cp16(gsrc + (size_t)(kt + 2) * 1024, (kt & 1) ? ldst1 : ldst0);
    }
#else
    // fallback: register-staged synchronous copy, still double buffered
    v16bf stg = *reinterpret_cast<const v16bf*>(gsrc);
    v16bf a_cur = Af[0];
    for (int kt = 0; kt < KTILES; ++kt) {
        *reinterpret_cast<v16bf*>((kt & 1) ? ldst1 : ldst0) = stg;
        if (kt < KTILES - 1)
            stg = *reinterpret_cast<const v16bf*>(gsrc + (size_t)(kt + 1) * 1024);
        __syncthreads();

        const v16bf* Bl = reinterpret_cast<const v16bf*>(smem + (kt & 1) * 4096);
        v16bf bbA = Bl[0 * 32 + lane];
        v16bf bbB = Bl[1 * 32 + lane];

        v16bf a_use = a_cur;
        if (kt < KTILES - 1) a_cur = Af[(size_t)(kt + 1) * 32];

        acc[0] = __builtin_amdgcn_wmma_f32_16x16x32_bf16(false, a_use, false, bbA,
                                                         (short)0, acc[0], false, false);
        bbA = Bl[2 * 32 + lane];
        acc[1] = __builtin_amdgcn_wmma_f32_16x16x32_bf16(false, a_use, false, bbB,
                                                         (short)0, acc[1], false, false);
        bbB = Bl[3 * 32 + lane];
        acc[2] = __builtin_amdgcn_wmma_f32_16x16x32_bf16(false, a_use, false, bbA,
                                                         (short)0, acc[2], false, false);
        acc[3] = __builtin_amdgcn_wmma_f32_16x16x32_bf16(false, a_use, false, bbB,
                                                         (short)0, acc[3], false, false);
        __syncthreads();
    }
#endif

    float s     = scale[0];
    int   col   = lane & 15;
    int   rbase = rowtile * 16 + ((lane >> 4) << 3);
#pragma unroll
    for (int qq = 0; qq < 4; ++qq) {
        int ncol = (nh + qq) * 16 + col;
#pragma unroll
        for (int r = 0; r < 8; ++r) {
            out[(size_t)(rbase + r) * OUT_DIM + ncol] = acc[qq][r] * s;
        }
    }
}

// ---------------------------------------------------------------------------
extern "C" void kernel_launch(void* const* d_in, const int* in_sizes, int n_in,
                              void* d_out, int out_size, void* d_ws, size_t ws_size,
                              hipStream_t stream) {
    const float* x            = (const float*)d_in[0];   // (8192,128)
    const float* grid         = (const float*)d_in[1];   // (128,12)
    const float* spline_coeff = (const float*)d_in[2];   // (128,128,8)
    const float* base_weight  = (const float*)d_in[3];   // (128,128)
    const float* scale        = (const float*)d_in[4];   // (1,)
    float*       out          = (float*)d_out;           // (8192,128)

    unsigned short* Gpk = (unsigned short*)d_ws;                           // 294912 B
    unsigned short* Apk = (unsigned short*)((char*)d_ws + (512u * 1024u)); // 18.9 MB

    kan_features<<<(BATCH * IN_DIM) / 256, 256, 0, stream>>>(x, grid, Apk);
    kan_pack_w<<<(NTILES * KTILES * 32 * 16) / 256, 256, 0, stream>>>(
        spline_coeff, base_weight, Gpk);
    kan_gemm<<<dim3(64, 2), 256, 0, stream>>>(Apk, Gpk, out, scale);
    (void)in_sizes; (void)n_in; (void)out_size; (void)ws_size;
}
